// MistralMHA_36764920054210
// MI455X (gfx1250) — matrix operand order repaired and checked by reading, not compiled, not run on hardware
//
#include <hip/hip_runtime.h>

// ---------------------------------------------------------------------------
// Mistral MHA forward for gfx1250 (MI455X): bf16 WMMA everywhere.
//   B=2, S=2048, DIM=4096, H=32, KVH=8, D=128. WINDOW==S -> pure causal.
// ---------------------------------------------------------------------------

typedef __attribute__((ext_vector_type(16))) __bf16 v16bf;
typedef __attribute__((ext_vector_type(8)))  __bf16 v8bf;
typedef __attribute__((ext_vector_type(8)))  float  v8f;
typedef __bf16 bf16;

#define SEQL 2048
#define NH   32
#define NKV  8
#define HD   128
#define DMODEL 4096
#define BSZ  2

// ---- bf16 helpers (round-to-nearest-even) ---------------------------------
__device__ __forceinline__ unsigned short f2bfbits(float f) {
  unsigned u = __builtin_bit_cast(unsigned, f);
  unsigned r = u + 0x7FFFu + ((u >> 16) & 1u);
  return (unsigned short)(r >> 16);
}
__device__ __forceinline__ float bf2f(unsigned short b) {
  unsigned u = ((unsigned)b) << 16;
  return __builtin_bit_cast(float, u);
}
__device__ __forceinline__ unsigned pk2(float a, float b) {
  return (unsigned)f2bfbits(a) | ((unsigned)f2bfbits(b) << 16);
}

// B-fragment load: 16 contiguous bf16 (lane holds K = lh*16 .. +15)
__device__ __forceinline__ v16bf ldB(const bf16* p) {
  return *(const v16bf*)p;
}
// A-fragment load (contiguous, matching B): lane half lh holds K chunks
//   lh=0: K 0..15 ; lh=1: K 16..31
__device__ __forceinline__ v16bf ldA(const bf16* rowp, int lh8) {
  v8bf lo = *(const v8bf*)(rowp + 2 * lh8);
  v8bf hi = *(const v8bf*)(rowp + 2 * lh8 + 8);
  return __builtin_shufflevector(lo, hi, 0,1,2,3,4,5,6,7,8,9,10,11,12,13,14,15);
}

// ---- f32 -> bf16 conversion (4 elems / thread) ----------------------------
__global__ __launch_bounds__(256) void cvt_f32_bf16(const float* __restrict__ s,
                                                    bf16* __restrict__ d, int n4) {
  int i = blockIdx.x * 256 + threadIdx.x;
  if (i >= n4) return;
  float4 v = ((const float4*)s)[i];
  uint2 o;
  o.x = pk2(v.x, v.y);
  o.y = pk2(v.z, v.w);
  ((uint2*)d)[i] = o;
}

// ---- RoPE (in-place on bf16, fp32 math), optional score-scale fold --------
__global__ __launch_bounds__(256) void rope_kernel(bf16* __restrict__ t,
                                                   const float* __restrict__ cosb,
                                                   const float* __restrict__ sinb,
                                                   int nheads, float scale) {
  int idx = blockIdx.x * 256 + threadIdx.x;
  int i = idx & 63;                       // pair index (HD/2 = 64)
  int h = (idx >> 6) % nheads;
  int s = (idx / (64 * nheads)) % SEQL;
  int b = idx / (64 * nheads * SEQL);
  if (b >= BSZ) return;
  bf16* p = t + (((size_t)(b * SEQL + s) * nheads + h) * HD) + 2 * i;
  unsigned pr = *(const unsigned*)p;
  float x0 = bf2f((unsigned short)(pr & 0xFFFFu));
  float x1 = bf2f((unsigned short)(pr >> 16));
  float c = cosb[s * 64 + i], sn = sinb[s * 64 + i];
  float r0 = (x0 * c - x1 * sn) * scale;
  float r1 = (x0 * sn + x1 * c) * scale;
  *(unsigned*)p = pk2(r0, r1);
}

// ---- NT GEMM: C[m,n] = sum_k A[m,k] * W[n,k]  (bf16 in, f32 accum) --------
// wave tile 32(M) x 64(N); mode 0: bf16 row-major, 1: V transposed store,
// mode 2: f32 row-major (final output).
__global__ __launch_bounds__(256) void gemm_nt_bf16(const bf16* __restrict__ A,
                                                    const bf16* __restrict__ W,
                                                    bf16* __restrict__ Cb,
                                                    float* __restrict__ Cf,
                                                    int M, int N, int K, int mode) {
  int wid = blockIdx.x * 8 + (threadIdx.x >> 5);
  int l = threadIdx.x & 31, lr = l & 15, lh = l >> 4;
  int nNT = N >> 6;
  int mt = wid / nNT, nt = wid % nNT;
  if (mt * 32 >= M) return;
  int m0 = mt * 32, n0 = nt * 64;
  int lh8 = lh * 8, lh16 = lh * 16;

  v8f acc[2][4];
  for (int a = 0; a < 2; ++a)
    for (int b = 0; b < 4; ++b)
      for (int j = 0; j < 8; ++j) acc[a][b][j] = 0.f;

  const bf16* ap = A + (size_t)(m0 + lr) * K;
  const bf16* wp = W + (size_t)(n0 + lr) * K;

  for (int k = 0; k < K; k += 32) {
    v16bf a0 = ldA(ap + k, lh8);
    v16bf a1 = ldA(ap + (size_t)16 * K + k, lh8);
    v16bf b0 = ldB(wp + k + lh16);
    v16bf b1 = ldB(wp + (size_t)16 * K + k + lh16);
    v16bf b2 = ldB(wp + (size_t)32 * K + k + lh16);
    v16bf b3 = ldB(wp + (size_t)48 * K + k + lh16);
    acc[0][0] = __builtin_amdgcn_wmma_f32_16x16x32_bf16(false, a0, false, b0, (short)0, acc[0][0], false, false);
    acc[0][1] = __builtin_amdgcn_wmma_f32_16x16x32_bf16(false, a0, false, b1, (short)0, acc[0][1], false, false);
    acc[0][2] = __builtin_amdgcn_wmma_f32_16x16x32_bf16(false, a0, false, b2, (short)0, acc[0][2], false, false);
    acc[0][3] = __builtin_amdgcn_wmma_f32_16x16x32_bf16(false, a0, false, b3, (short)0, acc[0][3], false, false);
    acc[1][0] = __builtin_amdgcn_wmma_f32_16x16x32_bf16(false, a1, false, b0, (short)0, acc[1][0], false, false);
    acc[1][1] = __builtin_amdgcn_wmma_f32_16x16x32_bf16(false, a1, false, b1, (short)0, acc[1][1], false, false);
    acc[1][2] = __builtin_amdgcn_wmma_f32_16x16x32_bf16(false, a1, false, b2, (short)0, acc[1][2], false, false);
    acc[1][3] = __builtin_amdgcn_wmma_f32_16x16x32_bf16(false, a1, false, b3, (short)0, acc[1][3], false, false);
  }

  for (int mi = 0; mi < 2; ++mi)
    for (int ni = 0; ni < 4; ++ni)
      for (int j = 0; j < 8; ++j) {
        int row = m0 + mi * 16 + j + 8 * lh;
        int col = n0 + ni * 16 + lr;
        float v = acc[mi][ni][j];
        if (mode == 0) {
          Cb[(size_t)row * N + col] = __builtin_bit_cast(bf16, f2bfbits(v));
        } else if (mode == 2) {
          Cf[(size_t)row * N + col] = v;
        } else {  // V transposed: vT[b][kvh][d][s]
          int b = row >> 11, s = row & 2047;      // S = 2048
          int kvh = col >> 7, d = col & 127;      // HD = 128
          Cb[((size_t)(b * NKV + kvh) * HD + d) * SEQL + s] =
              __builtin_bit_cast(bf16, f2bfbits(v));
        }
      }
}

// ---- Flash attention: one wave = 16 queries of one (b,h) ------------------
// Scores computed transposed (S^T = K x Q^T) so per-query softmax stats are
// per-lane scalars; P^T re-laid to a B-fragment through per-wave LDS.
__global__ __launch_bounds__(256) void attn_kernel(const bf16* __restrict__ q,
                                                   const bf16* __restrict__ k,
                                                   const bf16* __restrict__ vT,
                                                   bf16* __restrict__ o) {
  __shared__ __attribute__((aligned(64))) unsigned short sh[8 * 512];  // 8 waves x 16q x 32k bf16
  int wv = threadIdx.x >> 5, l = threadIdx.x & 31, lr = l & 15, lh = l >> 4;
  int lh8 = lh * 8, lh16 = lh * 16;
  int wid = blockIdx.x * 8 + wv;
  int qt = wid & 127;          // SEQ/16 = 128
  int h  = (wid >> 7) & 31;    // NH = 32
  int b  = wid >> 12;
  int q0 = qt * 16;
  int kvh = h >> 2;            // repeat factor 4
  unsigned short* myl = sh + wv * 512;

  // Preload Q as 4 B-fragments (d blocks of 32); 1/sqrt(128) already folded in.
  v16bf qf[4];
  const bf16* qp = q + ((size_t)(b * SEQL + q0 + lr) * NH + h) * HD + lh16;
  for (int db = 0; db < 4; ++db) qf[db] = ldB(qp + db * 32);

  v8f acc[8];
  for (int dt = 0; dt < 8; ++dt)
    for (int j = 0; j < 8; ++j) acc[dt][j] = 0.f;
  float m = -1e30f, ls = 0.f;

  int kend = q0 + 16;  // causal: keys 0 .. q0+15
  for (int kb = 0; kb < kend; kb += 32) {
    v8f st[2];
    for (int t = 0; t < 2; ++t) {
      v8f s;
      for (int j = 0; j < 8; ++j) s[j] = 0.f;
      const bf16* kp =
          k + ((size_t)(b * SEQL + kb + t * 16 + lr) * NKV + kvh) * HD;
      for (int db = 0; db < 4; ++db) {
        v16bf a = ldA(kp + db * 32, lh8);
        s = __builtin_amdgcn_wmma_f32_16x16x32_bf16(false, a, false, qf[db], (short)0, s, false, false);
      }
      int kt0 = kb + t * 16;
      if (kt0 >= q0) {  // tile touches/crosses diagonal -> mask key > query
        int qq = q0 + lr;
        for (int j = 0; j < 8; ++j) {
          int key = kt0 + j + 8 * lh;
          if (key > qq) s[j] = -1e9f;
        }
      }
      st[t] = s;
    }

    // online softmax (stats per query == per lane, replicated across halves)
    float mn = m;
    for (int t = 0; t < 2; ++t)
      for (int j = 0; j < 8; ++j) mn = fmaxf(mn, st[t][j]);
    mn = fmaxf(mn, __shfl_xor(mn, 16, 32));
    float corr = __expf(m - mn);
    m = mn;
    ls *= corr;
    for (int dt = 0; dt < 8; ++dt)
      for (int j = 0; j < 8; ++j) acc[dt][j] *= corr;

    float ps = 0.f;
    for (int t = 0; t < 2; ++t) {
      for (int j = 0; j < 8; ++j) {
        float p = __expf(st[t][j] - m);
        st[t][j] = p;
        ps += p;
      }
      // P^T -> LDS at [query][key] (keys j,j+1 consecutive per lane)
      for (int jp = 0; jp < 4; ++jp) {
        *(unsigned*)(myl + lr * 32 + t * 16 + lh * 8 + 2 * jp) =
            pk2(st[t][2 * jp], st[t][2 * jp + 1]);
      }
    }
    ps += __shfl_xor(ps, 16, 32);
    ls += ps;

    asm volatile("s_wait_dscnt 0" ::: "memory");  // LDS RAW (in-order DS + fence)

    // B-fragment of P^T: lane lr = query column, keys lh*16..+15 contiguous
    v16bf pb = *(const v16bf*)(myl + lr * 32 + lh16);

    // O^T += V^T x P^T  (A rows = d from transposed V, contiguous in keys)
    const bf16* vp =
        vT + ((size_t)(b * NKV + kvh) * HD + lr) * SEQL + kb;
    for (int dt = 0; dt < 8; ++dt) {
      v16bf a = ldA(vp + (size_t)dt * 16 * SEQL, lh8);
      acc[dt] = __builtin_amdgcn_wmma_f32_16x16x32_bf16(false, a, false, pb, (short)0, acc[dt], false, false);
    }
  }

  float inv = 1.0f / ls;
  // store O: attn[b][s=q0+lr][h*128 + d], d = dt*16 + 8*lh + j (j contiguous)
  bf16* op = o + ((size_t)(b * SEQL + q0 + lr)) * DMODEL + h * HD + 8 * lh;
  for (int dt = 0; dt < 8; ++dt) {
    uint4 u;
    u.x = pk2(acc[dt][0] * inv, acc[dt][1] * inv);
    u.y = pk2(acc[dt][2] * inv, acc[dt][3] * inv);
    u.z = pk2(acc[dt][4] * inv, acc[dt][5] * inv);
    u.w = pk2(acc[dt][6] * inv, acc[dt][7] * inv);
    *(uint4*)(op + dt * 16) = u;
  }
}

// ---------------------------------------------------------------------------
extern "C" void kernel_launch(void* const* d_in, const int* in_sizes, int n_in,
                              void* d_out, int out_size, void* d_ws, size_t ws_size,
                              hipStream_t stream) {
  const float* x  = (const float*)d_in[0];
  const float* fc = (const float*)d_in[1];
  const float* fs = (const float*)d_in[2];
  // d_in[3] positions, d_in[4] mask: unused (WINDOW==SEQ -> pure causal)
  const float* wq = (const float*)d_in[5];
  const float* wk = (const float*)d_in[6];
  const float* wv = (const float*)d_in[7];
  const float* wo = (const float*)d_in[8];
  float* out = (float*)d_out;

  char* ws = (char*)d_ws;
  size_t off = 0;
  auto alloc = [&](size_t elems) -> bf16* {
    bf16* p = (bf16*)(ws + off);
    off += ((elems * sizeof(bf16) + 255) & ~(size_t)255);
    return p;
  };
  const size_t XE  = (size_t)BSZ * SEQL * DMODEL;   // 16.78M
  const size_t WQE = (size_t)DMODEL * DMODEL;
  const size_t WKE = (size_t)NKV * HD * DMODEL;
  const size_t KVE = (size_t)BSZ * SEQL * NKV * HD;

  bf16* xb  = alloc(XE);
  bf16* wqb = alloc(WQE);
  bf16* wkb = alloc(WKE);
  bf16* wvb = alloc(WKE);
  bf16* wob = alloc(WQE);
  bf16* qb  = alloc(XE);
  bf16* kbf = alloc(KVE);
  bf16* vtb = alloc(KVE);
  bf16* ab  = alloc(XE);
  (void)ws_size; (void)in_sizes; (void)n_in; (void)out_size;

  cvt_f32_bf16<<<(int)(XE / 1024),  256, 0, stream>>>(x,  xb,  (int)(XE / 4));
  cvt_f32_bf16<<<(int)(WQE / 1024), 256, 0, stream>>>(wq, wqb, (int)(WQE / 4));
  cvt_f32_bf16<<<(int)(WKE / 1024), 256, 0, stream>>>(wk, wkb, (int)(WKE / 4));
  cvt_f32_bf16<<<(int)(WKE / 1024), 256, 0, stream>>>(wv, wvb, (int)(WKE / 4));
  cvt_f32_bf16<<<(int)(WQE / 1024), 256, 0, stream>>>(wo, wob, (int)(WQE / 4));

  const int M = BSZ * SEQL;  // 4096
  // Q projection: 4096x4096x4096
  gemm_nt_bf16<<<(M / 32) * (DMODEL / 64) / 8, 256, 0, stream>>>(
      xb, wqb, qb, nullptr, M, DMODEL, DMODEL, 0);
  // K projection: N=1024
  gemm_nt_bf16<<<(M / 32) * ((NKV * HD) / 64) / 8, 256, 0, stream>>>(
      xb, wkb, kbf, nullptr, M, NKV * HD, DMODEL, 0);
  // V projection, stored transposed [B][KVH][D][S]
  gemm_nt_bf16<<<(M / 32) * ((NKV * HD) / 64) / 8, 256, 0, stream>>>(
      xb, wvb, vtb, nullptr, M, NKV * HD, DMODEL, 1);

  const float qscale = 0.08838834764831845f;  // 1/sqrt(128), folded into Q
  rope_kernel<<<(BSZ * SEQL * NH * 64) / 256, 256, 0, stream>>>(qb, fc, fs, NH, qscale);
  rope_kernel<<<(BSZ * SEQL * NKV * 64) / 256, 256, 0, stream>>>(kbf, fc, fs, NKV, 1.0f);

  // Attention: 2*32*128 waves / 8 per block
  attn_kernel<<<(BSZ * NH * (SEQL / 16)) / 8, 256, 0, stream>>>(qb, kbf, vtb, ab);

  // Output projection -> f32 d_out
  gemm_nt_bf16<<<(M / 32) * (DMODEL / 64) / 8, 256, 0, stream>>>(
      ab, wob, nullptr, out, M, DMODEL, DMODEL, 2);
}
